// Transformer_20100446946063
// MI455X (gfx1250) — compile-verified, matrix operand order checked by reading, not run on hardware
//
#include <hip/hip_runtime.h>
#include <hip/hip_bf16.h>

// MI455X (gfx1250) fused tiny-attention transformer block.
// B=8192 batches, M=8, C=512, H=8, dq=16, dv=64.
// Strategy: bf16 WMMA (v_wmma_f32_16x16x32_bf16) for the three projections
// (they carry ~all 52 GFLOP), VALU for the tiny 8x8 softmax, and the
// beta@V / gamma / bias / residual fused in registers (h never materialized).

#define HNUM   8
#define CC     512
#define NB     8            // batches per workgroup
#define ROWS   64           // NB * 8 rows of x per workgroup
#define XSTR   520          // bf16 elems per LDS x row (512 + 8 pad)
#define FSTR   132          // f32 elems per LDS f/g row (128 + 4 pad)
#define KSTEPS 16           // 512 / 32
#define QPK_OFF 0           // ushort offsets into packed-weight workspace
#define KPK_OFF 65536       // 128 frags * 512 ushorts
#define VPK_OFF 131072      // Q + K regions

typedef __attribute__((ext_vector_type(16))) __bf16 v16bf;
typedef __attribute__((ext_vector_type(8)))  float  v8f;

union Frag16 { uint4 q[2]; v16bf v; };

__device__ __forceinline__ unsigned short f2bf(float f) {
  unsigned u = __float_as_uint(f);
  u += 0x7fffu + ((u >> 16) & 1u);        // round to nearest even
  return (unsigned short)(u >> 16);
}

// ---------------------------------------------------------------------------
// Kernel 1: convert Wq/Wk/Wv fp32 -> bf16 in WMMA-B-fragment-native layout.
// Fragment (ntile, kstep) = 1024B: [32 lanes x 16B (K lo chunk)] then
// [32 lanes x 16B (K hi chunk)]. Wave fragment load = 2 coalesced b128 loads.
// Lane L<16 : n = nt*16+L,    K chunks {k0..k0+7},  {k0+16..k0+23}
// Lane L>=16: n = nt*16+L-16, K chunks {k0+8..k0+15},{k0+24..k0+31}
// (matches CDNA5 ISA 16-bit 32x16 B-matrix layout)
// ---------------------------------------------------------------------------
__global__ void pack_weights_kernel(const float* __restrict__ Wq,
                                    const float* __restrict__ Wk,
                                    const float* __restrict__ Wv,
                                    unsigned short* __restrict__ ws) {
  int g = blockIdx.x * blockDim.x + threadIdx.x;   // 0 .. 24575
  int frag = g >> 5;
  int lane = g & 31;
  const float* W;
  unsigned short* dst;
  int fl;
  if (frag < 128)      { W = Wq; fl = frag;       dst = ws + QPK_OFF; }
  else if (frag < 256) { W = Wk; fl = frag - 128; dst = ws + KPK_OFF; }
  else                 { W = Wv; fl = frag - 256; dst = ws + VPK_OFF; }
  int ntile = fl >> 4;
  int kstep = fl & 15;
  int n  = ntile * 16 + (lane & 15);
  int k0 = kstep * 32 + ((lane >> 4) << 3);
  const float* src = W + (size_t)n * CC;
  unsigned short* o0 = dst + (size_t)fl * 512 + lane * 8;  // lo chunk
  unsigned short* o1 = o0 + 256;                           // hi chunk (+512B)
#pragma unroll
  for (int j = 0; j < 8; ++j) {
    o0[j] = f2bf(src[k0 + j]);
    o1[j] = f2bf(src[k0 + 16 + j]);
  }
}

// ---------------------------------------------------------------------------
// Kernel 2: fused attention block. One 256-thread workgroup per 8 batches.
// ---------------------------------------------------------------------------
__global__ __launch_bounds__(256)
void attn_fused_kernel(const float* __restrict__ x,
                       const float* __restrict__ bq,
                       const float* __restrict__ bk,
                       const float* __restrict__ bv,
                       const float* __restrict__ gamma,
                       const int*   __restrict__ mask,
                       const unsigned short* __restrict__ wpack,
                       float* __restrict__ out,
                       float* __restrict__ beta_out) {
  __shared__ unsigned short lds_x[ROWS * XSTR];        // 66,560 B (bf16 x)
  __shared__ float lds_f[ROWS * FSTR];                 // 33,792 B (Q proj)
  __shared__ float lds_g[ROWS * FSTR];                 // 33,792 B (K proj)
  __shared__ float lds_beta[NB * HNUM * 8 * 8];        // 16,384 B

  const int t    = threadIdx.x;
  const int wave = t >> 5;
  const int lane = t & 31;
  const size_t xbase = (size_t)blockIdx.x * (ROWS * CC);
  const float gm = gamma[0];

  // ---- Phase A: x fp32 -> bf16 into LDS (coalesced float4 loads) ----
#pragma unroll 1
  for (int i = 0; i < 32; ++i) {
    int e4 = t + i * 256;                       // float4 index, 0..8191
    float4 v = ((const float4*)(x + xbase))[e4];
    int e   = e4 << 2;
    int row = e >> 9;
    int col = e & 511;
    uint2 u;
    u.x = (unsigned)f2bf(v.x) | ((unsigned)f2bf(v.y) << 16);
    u.y = (unsigned)f2bf(v.z) | ((unsigned)f2bf(v.w) << 16);
    *(uint2*)&lds_x[row * XSTR + col] = u;
  }
  __syncthreads();

  // ---- Phase B: Q and K projections via WMMA bf16 ----
  // 64 tiles: [0,32) = Q (mtile 0..3 x ntile 0..7), [32,64) = K.
  const int arow    = (lane & 15);              // + mtile*16 later
  const int koffsel = (lane >> 4) << 3;         // upper half-wave: K+8
#pragma unroll 1
  for (int ti = 0; ti < 8; ++ti) {
    int tile  = wave * 8 + ti;
    int isK   = tile >= 32;
    int tt    = tile & 31;
    int mtile = tt >> 3;
    int ntile = tt & 7;
    const unsigned short* wb =
        wpack + (isK ? KPK_OFF : QPK_OFF) + ntile * (16 * 512);
    const int rbaseA = mtile * 16 + arow;
    v8f acc = {0.f, 0.f, 0.f, 0.f, 0.f, 0.f, 0.f, 0.f};
#pragma unroll 4
    for (int k = 0; k < KSTEPS; ++k) {
      Frag16 af, bfv;
      const unsigned short* ap = &lds_x[rbaseA * XSTR + k * 32 + koffsel];
      af.q[0] = *(const uint4*)(ap);
      af.q[1] = *(const uint4*)(ap + 16);
      const uint4* wq4 = (const uint4*)(wb + k * 512);
      bfv.q[0] = wq4[lane];
      bfv.q[1] = wq4[32 + lane];
      acc = __builtin_amdgcn_wmma_f32_16x16x32_bf16(
          false, af.v, false, bfv.v, (short)0, acc, false, false);
    }
    int c = ntile * 16 + (lane & 15);
    float bias = isK ? bk[c] : bq[c];
    float* dstf = isK ? lds_g : lds_f;
    int rbase = mtile * 16 + ((lane >> 4) << 3);
#pragma unroll
    for (int i = 0; i < 8; ++i)
      dstf[(rbase + i) * FSTR + c] = acc[i] + bias;
  }
  __syncthreads();

  // ---- Phase C: scores + softmax + masked renorm -> beta (LDS + global) ----
#pragma unroll 1
  for (int s = t; s < NB * HNUM * 8; s += 256) {     // 512 rows, 2 per thread
    int b = s >> 6;
    int h = (s >> 3) & 7;
    int m = s & 7;
    const float* fr = &lds_f[(b * 8 + m) * FSTR + h * 16];
    float sc[8];
#pragma unroll
    for (int n = 0; n < 8; ++n) {
      const float* gr = &lds_g[(b * 8 + n) * FSTR + h * 16];
      float d = 0.f;
#pragma unroll
      for (int dd = 0; dd < 16; ++dd) d += fr[dd] * gr[dd];
      sc[n] = d * 0.25f;                         // 1/sqrt(16)
    }
    float mx = sc[0];
#pragma unroll
    for (int n = 1; n < 8; ++n) mx = fmaxf(mx, sc[n]);
    float e[8], sum = 0.f;
#pragma unroll
    for (int n = 0; n < 8; ++n) { e[n] = __expf(sc[n] - mx); sum += e[n]; }
    float inv = 1.f / sum;
    int gb = blockIdx.x * NB + b;
    float bb[8], s2 = 0.f;
#pragma unroll
    for (int n = 0; n < 8; ++n) {
      float mk = (float)mask[gb * 8 + n];
      bb[n] = e[n] * inv * mk;
      s2 += bb[n];
    }
    float inv2 = 1.f / (s2 + 1e-12f);
#pragma unroll
    for (int n = 0; n < 8; ++n) {
      float v = bb[n] * inv2;
      lds_beta[((b * 8 + h) * 8 + m) * 8 + n] = v;
      beta_out[(((size_t)gb * HNUM + h) * 8 + m) * 8 + n] = v;
    }
  }
  __syncthreads();

  // ---- Phase D: V projection via WMMA + fused beta@h + gamma + residual ----
  // 128 tiles: mtile 0..3 (16 rows = 2 batches), ntile 0..31 (16 channels).
#pragma unroll 1
  for (int tv = wave; tv < 128; tv += 8) {
    int mtile = tv >> 5;
    int ntile = tv & 31;
    const unsigned short* wb = wpack + VPK_OFF + ntile * (16 * 512);
    const int rbaseA = mtile * 16 + arow;
    v8f acc = {0.f, 0.f, 0.f, 0.f, 0.f, 0.f, 0.f, 0.f};
#pragma unroll 4
    for (int k = 0; k < KSTEPS; ++k) {
      Frag16 af, bfv;
      const unsigned short* ap = &lds_x[rbaseA * XSTR + k * 32 + koffsel];
      af.q[0] = *(const uint4*)(ap);
      af.q[1] = *(const uint4*)(ap + 16);
      const uint4* wq4 = (const uint4*)(wb + k * 512);
      bfv.q[0] = wq4[lane];
      bfv.q[1] = wq4[32 + lane];
      acc = __builtin_amdgcn_wmma_f32_16x16x32_bf16(
          false, af.v, false, bfv.v, (short)0, acc, false, false);
    }
    int c    = ntile * 16 + (lane & 15);     // output channel 0..511
    int head = c >> 6;                       // dv = 64
    int lb   = (mtile << 1) + (lane >> 4);   // local batch 0..7
    float bvv = bv[c];
    float hv[8];
#pragma unroll
    for (int i = 0; i < 8; ++i) hv[i] = acc[i] + bvv;   // h[n=0..7, c]
    const float* bet = &lds_beta[(lb * 8 + head) * 64];
    size_t ob = ((size_t)blockIdx.x * NB + lb) * (8 * CC) + c;
#pragma unroll
    for (int m = 0; m < 8; ++m) {
      float o = 0.f;
#pragma unroll
      for (int n = 0; n < 8; ++n) o += bet[m * 8 + n] * hv[n];
      out[ob + m * CC] = gm * o + x[ob + m * CC];       // residual
    }
  }
}

extern "C" void kernel_launch(void* const* d_in, const int* in_sizes, int n_in,
                              void* d_out, int out_size, void* d_ws, size_t ws_size,
                              hipStream_t stream) {
  const float* x     = (const float*)d_in[0];
  const float* Wq    = (const float*)d_in[1];
  const float* bqv   = (const float*)d_in[2];
  const float* Wk    = (const float*)d_in[3];
  const float* bkv   = (const float*)d_in[4];
  const float* Wv    = (const float*)d_in[5];
  const float* bvv   = (const float*)d_in[6];
  const float* gamma = (const float*)d_in[7];
  const int*   mask  = (const int*)d_in[8];

  unsigned short* pack = (unsigned short*)d_ws;   // 786,432 bytes used
  float* out_p  = (float*)d_out;
  float* beta_p = out_p + (size_t)8192 * 8 * CC;  // tuple: (out, beta)

  // 768 fragments * 32 lanes = 24576 threads
  pack_weights_kernel<<<96, 256, 0, stream>>>(Wq, Wk, Wv, pack);
  attn_fused_kernel<<<8192 / NB, 256, 0, stream>>>(
      x, bqv, bkv, bvv, gamma, mask, pack, out_p, beta_p);
}
